// Model_76562087018931
// MI455X (gfx1250) — matrix operand order repaired and checked by reading, not compiled, not run on hardware
//
#include <hip/hip_runtime.h>
#include <hip/hip_bf16.h>

// DiffPool GNN for MI455X (gfx1250, wave32).
// Heavy level-1 batched GEMMs run on v_wmma_f32_16x16x32_f16 (f32 accumulate).
// an1 is materialized in f16 (128 MB -> resident in the 192 MB global L2).
// All WMMA operands use fragment-friendly layouts: B matrices are stored
// column-major (weights pre-transposed, XW/s1 written transposed by their
// producers), so every global->LDS stage and LDS->fragment gather is a
// contiguous b128 access (conflict-free 80B LDS strides). The k-loop is
// register software-pipelined: next tile's global loads are issued right
// after the barrier and their latency hides behind the current 8 WMMAs.
// All reductions are fp32 with deterministic two-stage reductions.
// Workspace requirement: ~280 MB.

typedef __attribute__((ext_vector_type(16))) _Float16 v16h;
typedef __attribute__((ext_vector_type(8)))  _Float16 v8h;
typedef __attribute__((ext_vector_type(8)))  float    v8f;

#define CDIV(a, b) (((a) + (b) - 1) / (b))

static __device__ __forceinline__ float wave_sum32(float v) {
#pragma unroll
  for (int o = 16; o > 0; o >>= 1) v += __shfl_down(v, o, 32);
  return v;
}

// ---------------------------------------------------------------------------
// Elementwise / prep kernels
// ---------------------------------------------------------------------------
__global__ void k_f32_to_f16(const float* __restrict__ s, _Float16* __restrict__ d, long long n) {
  long long i = (long long)blockIdx.x * blockDim.x + threadIdx.x;
  if (i < n) d[i] = (_Float16)s[i];
}

// transpose-convert a [K,N] f32 weight into [N,K] f16 (column-major storage)
__global__ void k_f32_to_f16_t(const float* __restrict__ s, _Float16* __restrict__ d,
                               int K, int N) {
  int i = blockIdx.x * blockDim.x + threadIdx.x;
  if (i >= K * N) return;
  int k = i / N, n = i % N;
  d[n * K + k] = (_Float16)s[i];
}

// one wave per adjacency row: dis[b,i] = rsqrt(max(rowsum + 1, 1e-12))
__global__ void k_row_dis(const float* __restrict__ adj, float* __restrict__ dis, int Nn) {
  int row  = blockIdx.x * 8 + (threadIdx.x >> 5);
  int lane = threadIdx.x & 31;
  const float* r = adj + (long long)row * Nn;
  float s = 0.0f;
  for (int j = lane; j < Nn; j += 32) s += r[j];
  s = wave_sum32(s);
  if (lane == 0) dis[row] = rsqrtf(fmaxf(s + 1.0f, 1e-12f));
}

// an1[b,i,j] = dis_i * dis_j * (adj + I)  in f16
__global__ void k_build_an16(const float* __restrict__ adj, const float* __restrict__ dis,
                             _Float16* __restrict__ an, int Nn) {
  long long idx = (long long)blockIdx.x * blockDim.x + threadIdx.x;
  long long NN = (long long)Nn * Nn;
  if (idx >= 64LL * NN) return;
  int b = (int)(idx / NN);
  long long rem = idx % NN;
  int r = (int)(rem / Nn), c = (int)(rem % Nn);
  float a = adj[idx] + (r == c ? 1.0f : 0.0f);
  an[idx] = (_Float16)(dis[b * Nn + r] * dis[b * Nn + c] * a);
}

// ---------------------------------------------------------------------------
// Main WMMA GEMM.  C[b] = A[b](MxK) @ B[b](KxN), f32 accum.
// B is ALWAYS stored column-major: logical B[k][n] = Bsrc[n*ldB + k].
// grid = (M/128, Nall/64, batch); block = 128 threads = 4 wave32 waves.
// Each wave computes a 32x64 strip (2 A-fragments x 4 B-tiles, 8 WMMA/k-step).
// A_F32: stage A from f32 (convert to f16 in LDS) -- used for raw adj.
// EPI: 0 = store f32, 1 = bias+relu -> f32, 2 = sum((adj - C)^2) -> partial[blk],
//      3 = store f16 row-major, 4 = store f16 transposed [b][col][row%1024]
//          (off = (row>>10)*sC + col*ldC + (row&1023))
// ---------------------------------------------------------------------------
template <int A_F32, int EPI>
__global__ __launch_bounds__(128) void k_wmma_gemm(
    const void* __restrict__ Asrc, const _Float16* __restrict__ Bsrc,
    const float* __restrict__ bias, const float* __restrict__ adjF,
    void* __restrict__ Cout,
    int K, int ldA, int ldB, int ldC,
    long long sA, long long sB, long long sC, long long sAdj) {
  // Row stride 40 halves = 80 B: multiple of 16 B (b128-aligned segments) and
  // 20 banks -> conflict-free across the 16 lanes of a fragment gather.
  __shared__ _Float16 Al[128][40];  // A block: 128 rows x 32 k
  __shared__ _Float16 Bt[64][40];   // B block, column-major: Bt[n][k]
  __shared__ float    red[128];

  const int tid = threadIdx.x;
  const int lane = tid & 31, wv = tid >> 5;
  const int laneHi = lane >> 4, lr = lane & 15;
  const int b = blockIdx.z;
  const long long blockM = (long long)blockIdx.x * 128;
  const long long blockN = (long long)blockIdx.y * 64;

  const _Float16* A16 = (const _Float16*)Asrc + (long long)b * sA;
  const float*    A32 = (const float*)Asrc    + (long long)b * sA;
  const _Float16* Bp  = Bsrc + (long long)b * sB;

  const v8f vzero = {0.f, 0.f, 0.f, 0.f, 0.f, 0.f, 0.f, 0.f};
  v8f acc[2][4];
#pragma unroll
  for (int f = 0; f < 2; ++f)
#pragma unroll
    for (int j = 0; j < 4; ++j) acc[f][j] = vzero;

  union V16 { v16h v; v8h h[2]; };

  // per-thread staging slices
  const int rA = tid;                      // A row owned by this thread
  const int nB = tid >> 1, k0 = (tid & 1) * 16;  // B column slice
  const _Float16* aBase16 = A16 + (blockM + rA) * (long long)ldA;
  const float*    aBase32 = A32 + (blockM + rA) * (long long)ldA;
  const _Float16* bBase   = Bp + (blockN + nB) * (long long)ldB + k0;

  // staging registers (software pipeline: hold tile kk while wmma runs on kk-32)
  v8h   ra[4];
  float ra32[32];
  v8h   rb0, rb1;

  // prologue: fetch tile kk = 0
  if (A_F32) {
#pragma unroll
    for (int j = 0; j < 32; ++j) ra32[j] = aBase32[j];
  } else {
    const v8h* s4 = (const v8h*)aBase16;
    ra[0] = s4[0]; ra[1] = s4[1]; ra[2] = s4[2]; ra[3] = s4[3];
  }
  rb0 = *(const v8h*)&bBase[0];
  rb1 = *(const v8h*)&bBase[8];

  for (int kk = 0; kk < K; kk += 32) {
    // ---- store staged registers to LDS ----
    if (A_F32) {
#pragma unroll
      for (int j = 0; j < 32; ++j) Al[rA][j] = (_Float16)ra32[j];
    } else {
      *(v8h*)&Al[rA][0]  = ra[0];
      *(v8h*)&Al[rA][8]  = ra[1];
      *(v8h*)&Al[rA][16] = ra[2];
      *(v8h*)&Al[rA][24] = ra[3];
    }
    *(v8h*)&Bt[nB][k0]     = rb0;
    *(v8h*)&Bt[nB][k0 + 8] = rb1;
    __syncthreads();

    // ---- issue next tile's global loads (wrapped addr on last iter; no
    //      branches). Latency hides behind the 8 WMMAs below. ----
    {
      int kn = (kk + 32 < K) ? (kk + 32) : 0;
      if (A_F32) {
        const float* s = aBase32 + kn;
#pragma unroll
        for (int j = 0; j < 32; ++j) ra32[j] = s[j];
      } else {
        const v8h* s4 = (const v8h*)(aBase16 + kn);
        ra[0] = s4[0]; ra[1] = s4[1]; ra[2] = s4[2]; ra[3] = s4[3];
      }
      rb0 = *(const v8h*)&bBase[kn];
      rb1 = *(const v8h*)&bBase[kn + 8];
    }

    // ---- gather fragments (ISA 7.12.2 layouts), 2x ds_load_b128 each ----
    v16h af[2];
#pragma unroll
    for (int f = 0; f < 2; ++f) {
      int arow = wv * 32 + f * 16 + lr;
      V16 a;
      a.h[0] = *(const v8h*)&Al[arow][laneHi * 8];        // k = laneHi*8 .. +7
      a.h[1] = *(const v8h*)&Al[arow][16 + laneHi * 8];   // k = 16+laneHi*8 ..
      af[f] = a.v;
    }
#pragma unroll
    for (int j = 0; j < 4; ++j) {
      V16 bb;
      bb.h[0] = *(const v8h*)&Bt[j * 16 + lr][laneHi * 8];       // k = laneHi*8 .. +7
      bb.h[1] = *(const v8h*)&Bt[j * 16 + lr][16 + laneHi * 8];
      v16h bf = bb.v;
#pragma unroll
      for (int f = 0; f < 2; ++f)
        acc[f][j] = __builtin_amdgcn_wmma_f32_16x16x32_f16(
            false, af[f], false, bf, (short)0, acc[f][j], false, false);
    }
    __syncthreads();
  }

  // ---- epilogue ----
  if (EPI == 2) {
    const float* adjp = adjF + (long long)b * sAdj;
    float ls = 0.0f;
#pragma unroll
    for (int f = 0; f < 2; ++f)
#pragma unroll
      for (int j = 0; j < 4; ++j) {
        long long col = blockN + j * 16 + lr;
#pragma unroll
        for (int v = 0; v < 8; ++v) {
          long long row = blockM + wv * 32 + f * 16 + v + laneHi * 8;
          float r = adjp[row * (long long)ldC + col] - acc[f][j][v];
          ls += r * r;
        }
      }
    red[tid] = ls;
    __syncthreads();
    if (tid < 64) red[tid] += red[tid + 64];
    __syncthreads();
    if (tid < 32) red[tid] += red[tid + 32];
    __syncthreads();
    if (tid == 0) {
      float t = 0.0f;
      for (int i = 0; i < 32; ++i) t += red[i];
      int pid = blockIdx.x + gridDim.x * (blockIdx.y + gridDim.y * blockIdx.z);
      ((float*)Cout)[pid] = t;   // deterministic block partial
    }
  } else {
#pragma unroll
    for (int f = 0; f < 2; ++f)
#pragma unroll
      for (int j = 0; j < 4; ++j) {
        long long col = blockN + j * 16 + lr;
#pragma unroll
        for (int v = 0; v < 8; ++v) {
          long long row = blockM + wv * 32 + f * 16 + v + laneHi * 8;
          float val = acc[f][j][v];
          if (EPI == 1) { val += bias[col]; val = fmaxf(val, 0.0f); }
          if (EPI == 4) {
            long long off = (row >> 10) * sC + col * (long long)ldC + (row & 1023);
            ((_Float16*)Cout)[off] = (_Float16)val;
          } else {
            long long off = (long long)b * sC + row * (long long)ldC + col;
            if (EPI == 3) ((_Float16*)Cout)[off] = (_Float16)val;
            else          ((float*)Cout)[off]    = val;
          }
        }
      }
  }
}

// ---------------------------------------------------------------------------
// BatchNorm (training mode): deterministic two-stage moments over R rows x 64 ch
// ---------------------------------------------------------------------------
__global__ void k_bn_stats(const float* __restrict__ t, float* __restrict__ bnpart, int R) {
  __shared__ float sS[256], sQ[256];
  int tid = threadIdx.x;
  int c = tid & 63, g = tid >> 6;
  int rowEnd = blockIdx.x * 64 + 64;
  float s = 0.0f, q = 0.0f;
  for (int r = blockIdx.x * 64 + g; r < rowEnd; r += 4) {
    float v = t[(long long)r * 64 + c];
    s += v; q += v * v;
  }
  sS[tid] = s; sQ[tid] = q;
  __syncthreads();
  if (tid < 64) {
    s = sS[tid] + sS[tid + 64] + sS[tid + 128] + sS[tid + 192];
    q = sQ[tid] + sQ[tid + 64] + sQ[tid + 128] + sQ[tid + 192];
    bnpart[(long long)blockIdx.x * 128 + tid]      = s;
    bnpart[(long long)blockIdx.x * 128 + 64 + tid] = q;
  }
}

__global__ void k_bn_reduce(const float* __restrict__ bnpart, float* __restrict__ stats, int nb) {
  int tid = threadIdx.x;   // 128 threads: [0..63]=sum, [64..127]=sumsq
  if (tid < 128) {
    float s = 0.0f;
    for (int b = 0; b < nb; ++b) s += bnpart[(long long)b * 128 + tid];
    stats[tid] = s;
  }
}

__global__ void k_bn_apply(const float* __restrict__ t, const float* __restrict__ stats,
                           const float* __restrict__ gm, const float* __restrict__ bt,
                           float* __restrict__ of32, _Float16* __restrict__ of16, int R) {
  long long i = (long long)blockIdx.x * blockDim.x + threadIdx.x;
  if (i >= (long long)R * 64) return;
  int c = (int)(i & 63);
  float inv = 1.0f / (float)R;
  float m = stats[c] * inv;
  float var = stats[64 + c] * inv - m * m;
  float y = gm[c] * (t[i] - m) * rsqrtf(var + 1e-5f) + bt[c];
  if (of32) of32[i] = y;
  if (of16) of16[i] = (_Float16)y;
}

// ---------------------------------------------------------------------------
// Softmax over 64 channels + entropy partials (thread per row).
// Optionally also writes the f16 result transposed ([b][c][row%1024]) for use
// as a column-major WMMA B operand (writes coalesce across consecutive rows).
// ---------------------------------------------------------------------------
__global__ void k_softmax_ent(const float* __restrict__ X, float* __restrict__ sf32,
                              _Float16* __restrict__ sf16, _Float16* __restrict__ sf16t,
                              float* __restrict__ entpart, int R) {
  __shared__ float red[256];
  long long row = (long long)blockIdx.x * blockDim.x + threadIdx.x;
  float e = 0.0f;
  if (row < R) {
    const float* x = X + row * 64;
    float mx = -1e30f;
    for (int c = 0; c < 64; ++c) mx = fmaxf(mx, x[c]);
    float sum = 0.0f;
    for (int c = 0; c < 64; ++c) sum += expf(x[c] - mx);
    float inv = 1.0f / sum;
    long long tb = (row >> 10) * 65536 + (row & 1023);
    for (int c = 0; c < 64; ++c) {
      float s = expf(x[c] - mx) * inv;
      if (sf32)  sf32[row * 64 + c] = s;
      if (sf16)  sf16[row * 64 + c] = (_Float16)s;
      if (sf16t) sf16t[tb + (long long)c * 1024] = (_Float16)s;
      e += -s * logf(s + 1e-15f);
    }
  }
  red[threadIdx.x] = e;
  __syncthreads();
  for (int o = 128; o > 0; o >>= 1) {
    if (threadIdx.x < o) red[threadIdx.x] += red[threadIdx.x + o];
    __syncthreads();
  }
  if (threadIdx.x == 0) entpart[blockIdx.x] = red[0];
}

// Deterministic fixed-order reduction of block partials into one scalar.
__global__ void k_reduce_to(float* __restrict__ dst, const float* __restrict__ part, int n) {
  __shared__ float red[256];
  float s = 0.0f;
  for (int i = threadIdx.x; i < n; i += 256) s += part[i];
  red[threadIdx.x] = s;
  __syncthreads();
  if (threadIdx.x == 0) {
    float t = 0.0f;
    for (int i = 0; i < 256; ++i) t += red[i];
    dst[0] = t;
  }
}

// ---------------------------------------------------------------------------
// Small fp32 batched GEMMs for levels 2/3 (M, N <= 128, K <= 1024; trivial FLOPs)
// ---------------------------------------------------------------------------
// C[b,m,n] = relu?(sum_k A[b,m,k]*B[(b,)k,n] + bias[n])
__global__ void k_nbgemm_nn(const float* __restrict__ A, const float* __restrict__ Bm,
                            const float* __restrict__ bias, float* __restrict__ C,
                            int M, int N, int K, int bShared, int doRelu) {
  long long i = (long long)blockIdx.x * blockDim.x + threadIdx.x;
  if (i >= 64LL * M * N) return;
  int n = (int)(i % N);
  long long t = i / N;
  int m = (int)(t % M), b = (int)(t / M);
  const float* a  = A + ((long long)b * M + m) * K;
  const float* bp = bShared ? (Bm + n) : (Bm + (long long)b * K * N + n);
  float s = 0.0f;
  for (int k = 0; k < K; ++k) s += a[k] * bp[(long long)k * N];
  if (bias) s += bias[n];
  if (doRelu) s = fmaxf(s, 0.0f);
  C[i] = s;
}

// C[b,m,n] = sum_k A[b,k,m]*B[b,k,n]   (s^T @ X style)
__global__ void k_nbgemm_tn(const float* __restrict__ A, const float* __restrict__ Bm,
                            float* __restrict__ C, int M, int N, int K) {
  long long i = (long long)blockIdx.x * blockDim.x + threadIdx.x;
  if (i >= 64LL * M * N) return;
  int n = (int)(i % N);
  long long t = i / N;
  int m = (int)(t % M), b = (int)(t / M);
  const float* a  = A + (long long)b * K * M + m;
  const float* bp = Bm + (long long)b * K * N + n;
  float s = 0.0f;
  for (int k = 0; k < K; ++k) s += a[(long long)k * M] * bp[(long long)k * N];
  C[i] = s;
}

// GCN-normalize a [B,64,64] adjacency (one block of 64 threads per batch)
__global__ void k_norm_small(const float* __restrict__ a, float* __restrict__ an) {
  __shared__ float dsh[64];
  int b = blockIdx.x, t = threadIdx.x;
  const float* ab = a + (long long)b * 4096;
  float s = 0.0f;
  for (int j = 0; j < 64; ++j) s += ab[t * 64 + j];
  dsh[t] = rsqrtf(fmaxf(s + 1.0f, 1e-12f));
  __syncthreads();
  float di = dsh[t];
  float* ob = an + (long long)b * 4096;
  for (int j = 0; j < 64; ++j) {
    float v = ab[t * 64 + j] + (t == j ? 1.0f : 0.0f);
    ob[t * 64 + j] = di * dsh[j] * v;
  }
}

// link loss level 2: partial[blk] = sum (a2 - s2 s2^T)^2 over its elements
__global__ void k_loss_small(const float* __restrict__ a2, const float* __restrict__ s2,
                             float* __restrict__ part) {
  __shared__ float red[256];
  long long i = (long long)blockIdx.x * 256 + threadIdx.x;   // 64*64*64 exact
  int j = (int)(i & 63), ii = (int)((i >> 6) & 63), b = (int)(i >> 12);
  const float* si = s2 + ((long long)b * 64 + ii) * 64;
  const float* sj = s2 + ((long long)b * 64 + j) * 64;
  float p = 0.0f;
  for (int k = 0; k < 64; ++k) p += si[k] * sj[k];
  float r = a2[i] - p;
  red[threadIdx.x] = r * r;
  __syncthreads();
  for (int o = 128; o > 0; o >>= 1) {
    if (threadIdx.x < o) red[threadIdx.x] += red[threadIdx.x + o];
    __syncthreads();
  }
  if (threadIdx.x == 0) part[blockIdx.x] = red[0];
}

__global__ void k_mean_nodes(const float* __restrict__ x, float* __restrict__ g) {
  int i = blockIdx.x * blockDim.x + threadIdx.x;   // B*64
  if (i >= 4096) return;
  int c = i & 63, b = i >> 6;
  const float* xb = x + (long long)b * 4096;
  float s = 0.0f;
  for (int k = 0; k < 64; ++k) s += xb[k * 64 + c];
  g[i] = s * (1.0f / 64.0f);
}

__global__ void k_lin(const float* __restrict__ X, const float* __restrict__ W,
                      const float* __restrict__ bias, float* __restrict__ O,
                      int M, int N, int K, int doRelu) {
  int i = blockIdx.x * blockDim.x + threadIdx.x;
  if (i >= M * N) return;
  int n = i % N, m = i / N;
  float s = 0.0f;
  for (int k = 0; k < K; ++k) s += X[m * K + k] * W[k * N + n];
  s += bias[n];
  if (doRelu) s = fmaxf(s, 0.0f);
  O[i] = s;
}

__global__ void k_head(const float* __restrict__ g1, const float* __restrict__ W2,
                       const float* __restrict__ b2, float* __restrict__ out) {
  int b = blockIdx.x * blockDim.x + threadIdx.x;
  if (b >= 64) return;
  float z[10];
  float mx = -1e30f;
  for (int n = 0; n < 10; ++n) {
    float s = b2[n];
    for (int k = 0; k < 128; ++k) s += g1[b * 128 + k] * W2[k * 10 + n];
    z[n] = s;
    mx = fmaxf(mx, s);
  }
  float sum = 0.0f;
  for (int n = 0; n < 10; ++n) sum += expf(z[n] - mx);
  float lse = logf(sum) + mx;
  for (int n = 0; n < 10; ++n) out[b * 10 + n] = z[n] - lse;
}

__global__ void k_final_loss(const float* __restrict__ scal, float* __restrict__ out) {
  if (threadIdx.x == 0 && blockIdx.x == 0) {
    float link1 = sqrtf(scal[0]) / 67108864.0f;   // B*N*N
    float link2 = sqrtf(scal[2]) / 262144.0f;     // B*64*64
    float ent1  = scal[1] / 65536.0f;             // B*N
    float ent2  = scal[3] / 4096.0f;              // B*64
    out[640] = link1 + link2 + ent1 + ent2;
  }
}

// ---------------------------------------------------------------------------
// Orchestration
// ---------------------------------------------------------------------------
extern "C" void kernel_launch(void* const* d_in, const int* in_sizes, int n_in,
                              void* d_out, int out_size, void* d_ws, size_t ws_size,
                              hipStream_t stream) {
  (void)in_sizes; (void)n_in; (void)out_size; (void)ws_size;

  // inputs (setup_inputs dict order, params dict insertion order)
  const float* x     = (const float*)d_in[0];
  const float* adj   = (const float*)d_in[1];
  const float* g1eW  = (const float*)d_in[2];
  const float* g1eb  = (const float*)d_in[3];
  const float* g2eW  = (const float*)d_in[4];
  const float* g2eb  = (const float*)d_in[5];
  const float* g3eW  = (const float*)d_in[6];
  const float* g3eb  = (const float*)d_in[7];
  // p1 block: 8..19, p2 block: 20..31  (c{i}_W, c{i}_b, bn{i}_g, bn{i}_b)
  const float* lin1W = (const float*)d_in[32];
  const float* lin1b = (const float*)d_in[33];
  const float* lin2W = (const float*)d_in[34];
  const float* lin2b = (const float*)d_in[35];
  float* out = (float*)d_out;

  // workspace layout (bytes)
  char* W = (char*)d_ws;
  size_t o = 0;
  auto take = [&](size_t bytes) { size_t r = o; o += (bytes + 255) & ~(size_t)255; return r; };
  _Float16* an16  = (_Float16*)(W + take(134217728)); // [B,1024,1024] f16
  _Float16* xf16  = (_Float16*)(W + take(16777216));  // [B,1024,128]  f16
  _Float16* xwt16 = (_Float16*)(W + take(8388608));   // XW transposed [B,64,1024] f16
  _Float16* h16   = (_Float16*)(W + take(8388608));   // [B,1024,64]   f16
  float*    t32   = (float*)(W + take(16777216));     // [B,1024,64]   f32
  float*    h32   = (float*)(W + take(16777216));     // [B,1024,64]   f32
  _Float16* s116  = (_Float16*)(W + take(8388608));   // s1 f16 row-major
  _Float16* s1t16 = (_Float16*)(W + take(8388608));   // s1 f16 transposed [B,64,1024]
  float*    s132  = (float*)(W + take(16777216));     // s1 f32
  float*    x132  = (float*)(W + take(16777216));     // embed1 out f32
  float*    T32   = (float*)(W + take(16777216));     // adj@s1 f32
  float*    dis1  = (float*)(W + take(262144));       // [B,1024]
  _Float16* w16   = (_Float16*)(W + take(65536));     // transposed f16 weights
  float*    stats = (float*)(W + take(1024));         // 128 floats
  float*    scal  = (float*)(W + take(256));          // ss1, ent1, ss2, ent2
  float*    part  = (float*)(W + take(65536));        // up to 16384 partials
  float*    bnp   = (float*)(W + take(524288));       // 1024 blocks * 128
  auto sbuf = [&]() { return (float*)(W + take(1048576)); };  // [B,64,64] f32
  float *x2 = sbuf(), *a2 = sbuf(), *an2 = sbuf(), *sxw = sbuf(), *st = sbuf(),
        *sh = sbuf(), *s2 = sbuf(), *x2e = sbuf(), *tta = sbuf(), *x3 = sbuf(),
        *a3 = sbuf(), *an3 = sbuf(), *x3e = sbuf();
  float* gpool = (float*)(W + take(16384));           // [B,64]
  float* g1buf = (float*)(W + take(32768));           // [B,128]

  _Float16* w16c1  = w16;            // [64,128] (transposed 128x64)
  _Float16* w16c2  = w16 + 8192;     // [64,64]
  _Float16* w16c3  = w16 + 12288;    // [64,64]
  _Float16* w16g1e = w16 + 16384;    // [64,128]

  // ---- prep ----
  k_f32_to_f16<<<CDIV(8388608LL, 256), 256, 0, stream>>>(x, xf16, 8388608LL);
  k_f32_to_f16_t<<<32, 256, 0, stream>>>((const float*)d_in[8],  w16c1, 128, 64);
  k_f32_to_f16_t<<<16, 256, 0, stream>>>((const float*)d_in[12], w16c2, 64, 64);
  k_f32_to_f16_t<<<16, 256, 0, stream>>>((const float*)d_in[16], w16c3, 64, 64);
  k_f32_to_f16_t<<<32, 256, 0, stream>>>(g1eW, w16g1e, 128, 64);
  k_row_dis<<<8192, 256, 0, stream>>>(adj, dis1, 1024);
  k_build_an16<<<262144, 256, 0, stream>>>(adj, dis1, an16, 1024);

  // ---- level-1 pooling GNN: 3 x (GCN + BN) then softmax ----
  const _Float16* wlist[3] = {w16c1, w16c2, w16c3};
  for (int i = 0; i < 3; ++i) {
    const float* cb  = (const float*)d_in[9 + 4 * i];
    const float* bng = (const float*)d_in[10 + 4 * i];
    const float* bnb = (const float*)d_in[11 + 4 * i];
    int K = (i == 0) ? 128 : 64;
    const void* Ain = (i == 0) ? (const void*)xf16 : (const void*)h16;
    // XW = X @ W, stored transposed f16 (flat M = 65536)
    k_wmma_gemm<0, 4><<<dim3(512, 1, 1), 128, 0, stream>>>(
        Ain, wlist[i], nullptr, nullptr, xwt16, K, K, K, 1024, 0, 0, 65536LL, 0);
    // t = relu(an1 @ XW + b)  (batched, f32 out)
    k_wmma_gemm<0, 1><<<dim3(8, 1, 64), 128, 0, stream>>>(
        an16, xwt16, cb, nullptr, t32, 1024, 1024, 1024, 64, 1048576LL, 65536LL, 65536LL, 0);
    // BatchNorm (deterministic moments)
    k_bn_stats<<<1024, 256, 0, stream>>>(t32, bnp, 65536);
    k_bn_reduce<<<1, 128, 0, stream>>>(bnp, stats, 1024);
    k_bn_apply<<<16384, 256, 0, stream>>>(t32, stats, bng, bnb, h32, h16, 65536);
  }
  // s1 = softmax(h), entropy partials -> scal[1]
  k_softmax_ent<<<256, 256, 0, stream>>>(h32, s132, s116, s1t16, part, 65536);
  k_reduce_to<<<1, 256, 0, stream>>>(&scal[1], part, 256);

  // ---- level-1 embed: x1 = relu(an1 @ (x @ W_g1e) + b) ----
  k_wmma_gemm<0, 4><<<dim3(512, 1, 1), 128, 0, stream>>>(
      xf16, w16g1e, nullptr, nullptr, xwt16, 128, 128, 128, 1024, 0, 0, 65536LL, 0);
  k_wmma_gemm<0, 1><<<dim3(8, 1, 64), 128, 0, stream>>>(
      an16, xwt16, g1eb, nullptr, x132, 1024, 1024, 1024, 64, 1048576LL, 65536LL, 65536LL, 0);

  // ---- link loss 1: ss1 = sum((adj - s1 s1^T)^2)  (B = s1^T via row-major s1) ----
  k_wmma_gemm<0, 2><<<dim3(8, 16, 64), 128, 0, stream>>>(
      s116, s116, nullptr, adj, part, 64, 64, 64, 1024, 65536LL, 65536LL, 0, 1048576LL);
  k_reduce_to<<<1, 256, 0, stream>>>(&scal[0], part, 8192);

  // ---- coarsen level 1: T = adj @ s1 ; x2 = s1^T x1 ; a2 = s1^T T ----
  k_wmma_gemm<1, 0><<<dim3(8, 1, 64), 128, 0, stream>>>(
      adj, s1t16, nullptr, nullptr, T32, 1024, 1024, 1024, 64, 1048576LL, 65536LL, 65536LL, 0);
  k_nbgemm_tn<<<1024, 256, 0, stream>>>(s132, x132, x2, 64, 64, 1024);
  k_nbgemm_tn<<<1024, 256, 0, stream>>>(s132, T32, a2, 64, 64, 1024);

  // ---- level 2 (all fp32, tiny) ----
  k_norm_small<<<64, 64, 0, stream>>>(a2, an2);
  const float* xcur = x2;
  for (int i = 0; i < 3; ++i) {
    const float* cW  = (const float*)d_in[20 + 4 * i];
    const float* cb  = (const float*)d_in[21 + 4 * i];
    const float* bng = (const float*)d_in[22 + 4 * i];
    const float* bnb = (const float*)d_in[23 + 4 * i];
    k_nbgemm_nn<<<1024, 256, 0, stream>>>(xcur, cW, nullptr, sxw, 64, 64, 64, 1, 0);
    k_nbgemm_nn<<<1024, 256, 0, stream>>>(an2, sxw, cb, st, 64, 64, 64, 0, 1);
    k_bn_stats<<<64, 256, 0, stream>>>(st, bnp, 4096);
    k_bn_reduce<<<1, 128, 0, stream>>>(bnp, stats, 64);
    k_bn_apply<<<1024, 256, 0, stream>>>(st, stats, bng, bnb, sh, nullptr, 4096);
    xcur = sh;
  }
  k_softmax_ent<<<16, 256, 0, stream>>>(sh, s2, nullptr, nullptr, part, 4096);
  k_reduce_to<<<1, 256, 0, stream>>>(&scal[3], part, 16);

  // embed2: x2e = relu(an2 @ (x2 @ W_g2e) + b)
  k_nbgemm_nn<<<1024, 256, 0, stream>>>(x2, g2eW, nullptr, sxw, 64, 64, 64, 1, 0);
  k_nbgemm_nn<<<1024, 256, 0, stream>>>(an2, sxw, g2eb, x2e, 64, 64, 64, 0, 1);

  // link loss 2
  k_loss_small<<<1024, 256, 0, stream>>>(a2, s2, part);
  k_reduce_to<<<1, 256, 0, stream>>>(&scal[2], part, 1024);

  // coarsen level 2: x3 = s2^T x2e ; a3 = s2^T (a2 @ s2)
  k_nbgemm_tn<<<1024, 256, 0, stream>>>(s2, x2e, x3, 64, 64, 64);
  k_nbgemm_nn<<<1024, 256, 0, stream>>>(a2, s2, nullptr, tta, 64, 64, 64, 0, 0);
  k_nbgemm_tn<<<1024, 256, 0, stream>>>(s2, tta, a3, 64, 64, 64);

  // level 3 embed + head
  k_norm_small<<<64, 64, 0, stream>>>(a3, an3);
  k_nbgemm_nn<<<1024, 256, 0, stream>>>(x3, g3eW, nullptr, sxw, 64, 64, 64, 1, 0);
  k_nbgemm_nn<<<1024, 256, 0, stream>>>(an3, sxw, g3eb, x3e, 64, 64, 64, 0, 1);
  k_mean_nodes<<<16, 256, 0, stream>>>(x3e, gpool);
  k_lin<<<32, 256, 0, stream>>>(gpool, lin1W, lin1b, g1buf, 64, 128, 64, 1);
  k_head<<<1, 64, 0, stream>>>(g1buf, lin2W, lin2b, out);
  k_final_loss<<<1, 32, 0, stream>>>(scal, out);
}
